// Attention_34789235097938
// MI455X (gfx1250) — matrix operand order repaired and checked by reading, not compiled, not run on hardware
//
#include <hip/hip_runtime.h>
#include <hip/hip_bf16.h>

#define BATCH 32
#define SEQ   2048
#define EMB   1024            // K (embed)
#define ATT   1024            // N (attn)
#define ROWS_TOTAL (BATCH * SEQ)

#define ROWS_PER_WG 128       // 8 waves x 16-row M-tiles
#define TPB   256
#define NTILES (ATT / 16)     // 64 N-tiles
#define KSTEPS (EMB / 32)     // 32 WMMA steps per tile

typedef __bf16 v16bf __attribute__((ext_vector_type(16)));
typedef __bf16 v8bf  __attribute__((ext_vector_type(8)));
typedef __bf16 v4bf  __attribute__((ext_vector_type(4)));
typedef float  v8f   __attribute__((ext_vector_type(8)));

__device__ __forceinline__ float fast_tanh(float x) {
  // tanh(x) = 1 - 2/(exp(2x)+1), exp via v_exp_f32 (exp2), div via v_rcp_f32
  float e = __builtin_amdgcn_exp2f(x * 2.8853900817779268f); // 2*log2(e)
  return 1.0f - 2.0f * __builtin_amdgcn_rcpf(e + 1.0f);
}

// ---------------------------------------------------------------------------
// Kernel 1: scores[row] = sum_n tanh( (x @ W)[row,n] + b[n] ) * ctx[n]
// GEMM on v_wmma_f32_16x16x32_bf16; h never hits memory.
// LDS: A-tile 256KB + double-buffered W-tile 2x32KB = 320KB (full WGP LDS).
// W tile nt+1 is prefetched into registers while tile nt runs on the XDL.
// ---------------------------------------------------------------------------
__global__ __launch_bounds__(TPB) void scores_kernel(
    const float* __restrict__ x, const float* __restrict__ W,
    const float* __restrict__ bias, const float* __restrict__ ctx,
    float* __restrict__ scores) {
  extern __shared__ char smem[];
  __bf16* As  = (__bf16*)smem;                               // [128][1024] row-major
  __bf16* Ws0 = (__bf16*)(smem + ROWS_PER_WG * EMB * 2);     // [16][1024] col-major
  __bf16* Ws1 = Ws0 + 16 * EMB;                              // double buffer

  const int tid  = threadIdx.x;
  const int wave = tid >> 5;
  const int lane = tid & 31;
  const int half = lane >> 4;   // K-half select (this code's contiguous k = 16*half + s order)
  const int l16  = lane & 15;   // A: M row (both halves). B/C: N column.
  const int row0 = blockIdx.x * ROWS_PER_WG;

  // W-tile staging geometry: thread covers 4 consecutive N-columns of one K-row
  const int wc4 = (tid & 3) * 4;   // first of 4 columns
  const int wk  = tid >> 2;        // K-row base (0..63), stride 64 over 16 passes

  // ---- stage 128 rows of x as bf16 into LDS (coalesced float4 reads) ----
  {
    const float4* xg = (const float4*)(x + (size_t)row0 * EMB);
    v4bf* As4 = (v4bf*)As;
    for (int i = tid; i < ROWS_PER_WG * EMB / 4; i += TPB) {
      float4 v = xg[i];
      v4bf o;
      o[0] = (__bf16)v.x; o[1] = (__bf16)v.y; o[2] = (__bf16)v.z; o[3] = (__bf16)v.w;
      As4[i] = o;
    }
  }

  // ---- prefetch W tile 0 into registers, commit to buffer 0 ----
  float4 wreg[16];
  {
    const float* wp = W + (size_t)wk * ATT + 0 * 16 + wc4;
#pragma unroll
    for (int p = 0; p < 16; ++p)
      wreg[p] = *(const float4*)(wp + (size_t)p * 64 * ATT);
#pragma unroll
    for (int p = 0; p < 16; ++p) {
      const int k = wk + p * 64;
      Ws0[(wc4 + 0) * EMB + k] = (__bf16)wreg[p].x;
      Ws0[(wc4 + 1) * EMB + k] = (__bf16)wreg[p].y;
      Ws0[(wc4 + 2) * EMB + k] = (__bf16)wreg[p].z;
      Ws0[(wc4 + 3) * EMB + k] = (__bf16)wreg[p].w;
    }
  }
  __syncthreads(); // A-tile + W-tile 0 visible

  float sacc[8] = {0.f, 0.f, 0.f, 0.f, 0.f, 0.f, 0.f, 0.f};

  const __bf16* arow = As + (wave * 16 + l16) * EMB;

  for (int nt = 0; nt < NTILES; ++nt) {
    const int n0 = nt * 16;
    const __bf16* Wcur = (nt & 1) ? Ws1 : Ws0;
    __bf16* Wnxt       = (nt & 1) ? Ws0 : Ws1;

    // ---- issue global prefetch of W tile nt+1 (latency hidden by WMMA) ----
    if (nt + 1 < NTILES) {
      const float* wp = W + (size_t)wk * ATT + (nt + 1) * 16 + wc4;
#pragma unroll
      for (int p = 0; p < 16; ++p)
        wreg[p] = *(const float4*)(wp + (size_t)p * 64 * ATT);
    }

    // ---- K-loop: 32 x v_wmma_f32_16x16x32_bf16 ----
    v8f acc = {};
    const __bf16* bcol = Wcur + l16 * EMB;
#pragma unroll 4
    for (int ks = 0; ks < KSTEPS; ++ks) {
      const int k0 = ks * 32;
      // A 16x32 bf16 frag: lane(half) holds 16 contiguous K [k0+half*16,+16)
      v8bf a_lo = *(const v8bf*)(arow + k0 + half * 16);
      v8bf a_hi = *(const v8bf*)(arow + k0 + half * 16 + 8);
      v16bf a = __builtin_shufflevector(a_lo, a_hi,
                 0,1,2,3,4,5,6,7,8,9,10,11,12,13,14,15);
      // B 32x16 bf16 frag: lane holds 16 contiguous K at column l16
      v16bf b = *(const v16bf*)(bcol + k0 + half * 16);
      acc = __builtin_amdgcn_wmma_f32_16x16x32_bf16(
          false, a, false, b, (short)0, acc, false, false);
    }

    // ---- fused epilogue: tanh(+bias) * ctx, accumulate per-row partials ----
    const float bn = bias[n0 + l16];
    const float cn = ctx[n0 + l16];
#pragma unroll
    for (int r = 0; r < 8; ++r)
      sacc[r] += fast_tanh(acc[r] + bn) * cn;

    // ---- commit prefetched tile into the alternate buffer ----
    if (nt + 1 < NTILES) {
#pragma unroll
      for (int p = 0; p < 16; ++p) {
        const int k = wk + p * 64;
        Wnxt[(wc4 + 0) * EMB + k] = (__bf16)wreg[p].x;
        Wnxt[(wc4 + 1) * EMB + k] = (__bf16)wreg[p].y;
        Wnxt[(wc4 + 2) * EMB + k] = (__bf16)wreg[p].z;
        Wnxt[(wc4 + 3) * EMB + k] = (__bf16)wreg[p].w;
      }
    }
    __syncthreads();
  }

  // reduce over the 16 N-columns held across each 16-lane half (wave32)
#pragma unroll
  for (int m = 1; m < 16; m <<= 1)
#pragma unroll
    for (int r = 0; r < 8; ++r)
      sacc[r] += __shfl_xor(sacc[r], m, 32);

  if (l16 == 0) {
    const int rbase = row0 + wave * 16 + half * 8; // lanes 0..15: rows r; 16..31: rows r+8
#pragma unroll
    for (int r = 0; r < 8; ++r) scores[rbase + r] = sacc[r];
  }
}

// ---------------------------------------------------------------------------
// Kernel 2: softmax over seq dim, in place. One WG per batch.
// ---------------------------------------------------------------------------
__global__ __launch_bounds__(256) void softmax_kernel(float* __restrict__ sc) {
  __shared__ float red[256];
  const int b = blockIdx.x, tid = threadIdx.x;
  float* s = sc + (size_t)b * SEQ;
  const float L2E = 1.4426950408889634f;

  float m = -3.4e38f;
  for (int i = tid; i < SEQ; i += 256) m = fmaxf(m, s[i]);
  red[tid] = m; __syncthreads();
  for (int o = 128; o > 0; o >>= 1) {
    if (tid < o) red[tid] = fmaxf(red[tid], red[tid + o]);
    __syncthreads();
  }
  m = red[0]; __syncthreads();

  float sum = 0.f;
  for (int i = tid; i < SEQ; i += 256)
    sum += __builtin_amdgcn_exp2f((s[i] - m) * L2E);
  red[tid] = sum; __syncthreads();
  for (int o = 128; o > 0; o >>= 1) {
    if (tid < o) red[tid] += red[tid + o];
    __syncthreads();
  }
  const float invs = 1.0f / red[0];
  for (int i = tid; i < SEQ; i += 256)
    s[i] = __builtin_amdgcn_exp2f((s[i] - m) * L2E) * invs;
}

// ---------------------------------------------------------------------------
// Kernel 3: partial weighted pooling per (batch, seq-block). Deterministic
// (fixed-order accumulation, no float atomics). Fully coalesced float4 stream.
// ---------------------------------------------------------------------------
#define SBLK 256
#define NSB  (SEQ / SBLK)   // 8
__global__ __launch_bounds__(256) void pool_partial_kernel(
    const float* __restrict__ x, const float* __restrict__ w,
    float* __restrict__ part) {
  const int b = blockIdx.x / NSB, sb = blockIdx.x % NSB, tid = threadIdx.x;
  const float4* xb = (const float4*)(x + ((size_t)b * SEQ + sb * SBLK) * EMB);
  const float*  wb = w + (size_t)b * SEQ + sb * SBLK;
  float4 acc = {0.f, 0.f, 0.f, 0.f};
  for (int s = 0; s < SBLK; ++s) {
    const float ww = wb[s];
    float4 xv = xb[(size_t)s * (EMB / 4) + tid];
    acc.x += ww * xv.x; acc.y += ww * xv.y;
    acc.z += ww * xv.z; acc.w += ww * xv.w;
  }
  ((float4*)part)[(size_t)(b * NSB + sb) * (EMB / 4) + tid] = acc;
}

// ---------------------------------------------------------------------------
// Kernel 4: reduce the NSB partials -> out[b][e]
// ---------------------------------------------------------------------------
__global__ __launch_bounds__(256) void pool_reduce_kernel(
    const float* __restrict__ part, float* __restrict__ out) {
  const int idx = blockIdx.x * 256 + threadIdx.x; // over BATCH*EMB
  const int b = idx / EMB, e = idx % EMB;
  float s = 0.f;
#pragma unroll
  for (int sb = 0; sb < NSB; ++sb)
    s += part[(size_t)(b * NSB + sb) * EMB + e];
  out[idx] = s;
}

// ---------------------------------------------------------------------------
extern "C" void kernel_launch(void* const* d_in, const int* in_sizes, int n_in,
                              void* d_out, int out_size, void* d_ws, size_t ws_size,
                              hipStream_t stream) {
  const float* x    = (const float*)d_in[0];
  const float* W    = (const float*)d_in[1];
  const float* bias = (const float*)d_in[2];
  const float* ctx  = (const float*)d_in[3];
  float* out = (float*)d_out;

  float* scores = (float*)d_ws;                    // BATCH*SEQ floats (256 KB)
  float* part   = scores + (size_t)BATCH * SEQ;    // BATCH*NSB*EMB floats (1 MB)

  const size_t lds = (size_t)(ROWS_PER_WG * EMB + 2 * 16 * EMB) * sizeof(__bf16); // 320 KB

  scores_kernel<<<ROWS_TOTAL / ROWS_PER_WG, TPB, lds, stream>>>(x, W, bias, ctx, scores);
  softmax_kernel<<<BATCH, 256, 0, stream>>>(scores);
  pool_partial_kernel<<<BATCH * NSB, 256, 0, stream>>>(x, scores, part);
  pool_reduce_kernel<<<BATCH * EMB / 256, 256, 0, stream>>>(part, out);
}
